// MyModel_89223650607748
// MI455X (gfx1250) — matrix operand order repaired and checked by reading, not compiled, not run on hardware
//
#include <hip/hip_runtime.h>
#include <hip/hip_bf16.h>
#include <math.h>

typedef __attribute__((ext_vector_type(16))) _Float16 v16h;
typedef __attribute__((ext_vector_type(8)))  float    v8f;

#define H 256
#define T 32
#define MVOC 4096

// ---------------------------------------------------------------------------
// B-packing: convert fp32 B (optionally transposed access) into f16 laid out
// as WMMA 16-bit B fragments in this file's contiguous k order (32x16 per tile, 512 halves/tile):
//   Bp[(kTile*Nt + nTile)*512 + lane*16 + i]  <->  B[k, col]
//   col = nTile*16 + (lane&15),  k = kTile*32 + (lane>>4)*16 + i
// One thread per packed element; writes fully coalesced.
// ---------------------------------------------------------------------------
__global__ void pack_b(const float* __restrict__ B, _Float16* __restrict__ Bp,
                       int N, int ldb, int transB)
{
    const int gid   = blockIdx.x * 256 + threadIdx.x;
    const int i     = gid & 15;
    const int lane  = (gid >> 4) & 31;
    const int tile  = gid >> 9;
    const int Nt    = N >> 4;
    const int nTile = tile % Nt;
    const int kTile = tile / Nt;
    const int col   = nTile * 16 + (lane & 15);
    const int k     = kTile * 32 + (lane >> 4) * 16 + i;
    const float v   = transB ? B[(size_t)col * ldb + k]
                             : B[(size_t)k * ldb + col];
    Bp[gid] = (_Float16)v;
}

// ---------------------------------------------------------------------------
// WMMA GEMM with pre-packed B: C[M,N] = A[M,K] * B (+bias)(, relu)
//   A fp32 row-major (lda); B packed f16 fragments (see pack_b).
//   Wave owns one mTile x 4 consecutive nTiles (4 accumulators); A fragment
//   converted once per K-step and reused by 4 v_wmma_f32_16x16x32_f16.
// Block = 128 (4 waves). grid = (M/16, N/256).
// ---------------------------------------------------------------------------
__global__ void wmma_gemm_pk(const float* __restrict__ A,
                             const _Float16* __restrict__ Bpack,
                             const float* __restrict__ bias,
                             float* __restrict__ C,
                             int M, int N, int K,
                             int lda, int ldc, int doRelu)
{
    const int lane   = threadIdx.x & 31;
    const int wave   = threadIdx.x >> 5;
    const int mTile  = blockIdx.x;
    const int ntBase = (blockIdx.y * 4 + wave) * 4;      // first of 4 nTiles
    if (ntBase * 16 >= N) return;                         // wave-uniform

    const int Nt     = N >> 4;
    const int rowA   = mTile * 16 + (lane & 15);
    const int kbaseA = (lane >> 4) * 16;

    const float*    Ap = A + (size_t)rowA * lda + kbaseA;
    const _Float16* bp = Bpack + (size_t)ntBase * 512 + (size_t)lane * 16;
    const size_t    bStep = (size_t)Nt * 512;             // halves per kTile

    v8f acc0 = {}, acc1 = {}, acc2 = {}, acc3 = {};

    const int Kt = K >> 5;
#pragma unroll 2
    for (int kt = 0; kt < Kt; ++kt) {
        // A fragment: one contiguous run of 16 floats -> 4x b128 + cvt_pk
        const float4 x0 = *(const float4*)(Ap);
        const float4 x1 = *(const float4*)(Ap + 4);
        const float4 x2 = *(const float4*)(Ap + 8);
        const float4 x3 = *(const float4*)(Ap + 12);
        v16h a;
        a[0] = (_Float16)x0.x;  a[1] = (_Float16)x0.y;
        a[2] = (_Float16)x0.z;  a[3] = (_Float16)x0.w;
        a[4] = (_Float16)x1.x;  a[5] = (_Float16)x1.y;
        a[6] = (_Float16)x1.z;  a[7] = (_Float16)x1.w;
        a[8] = (_Float16)x2.x;  a[9] = (_Float16)x2.y;
        a[10] = (_Float16)x2.z; a[11] = (_Float16)x2.w;
        a[12] = (_Float16)x3.x; a[13] = (_Float16)x3.y;
        a[14] = (_Float16)x3.z; a[15] = (_Float16)x3.w;

        // 4 B fragments: contiguous 32B vectors, 512 halves apart
        const v16h b0 = *(const v16h*)(bp);
        const v16h b1 = *(const v16h*)(bp + 512);
        const v16h b2 = *(const v16h*)(bp + 1024);
        const v16h b3 = *(const v16h*)(bp + 1536);

        acc0 = __builtin_amdgcn_wmma_f32_16x16x32_f16(false, a, false, b0, (short)0, acc0, false, false);
        acc1 = __builtin_amdgcn_wmma_f32_16x16x32_f16(false, a, false, b1, (short)0, acc1, false, false);
        acc2 = __builtin_amdgcn_wmma_f32_16x16x32_f16(false, a, false, b2, (short)0, acc2, false, false);
        acc3 = __builtin_amdgcn_wmma_f32_16x16x32_f16(false, a, false, b3, (short)0, acc3, false, false);

        Ap += 32;
        bp += bStep;
    }

    // Epilogue: C/D layout -> lane n = (lane&15); rows m = 8*(lane>>4)+v
    const int mBase = mTile * 16 + 8 * (lane >> 4);
    v8f accs[4] = { acc0, acc1, acc2, acc3 };
#pragma unroll
    for (int t = 0; t < 4; ++t) {
        const int cn = (ntBase + t) * 16 + (lane & 15);
        const float bv = bias ? bias[cn] : 0.0f;
#pragma unroll
        for (int v = 0; v < 8; ++v) {
            float x = accs[t][v] + bv;
            if (doRelu) x = x > 0.0f ? x : 0.0f;
            C[(size_t)(mBase + v) * ldc + cn] = x;
        }
    }
}

// ---------------------------------------------------------------------------
// Embedding gather-sum: seq[t,h] = sum_l emb[codes[t,l], h]
// ---------------------------------------------------------------------------
__global__ void embed_sum(const int* __restrict__ dCodes, const int* __restrict__ pCodes,
                          const float* __restrict__ dEmb, const float* __restrict__ pEmb,
                          float* __restrict__ seqD, float* __restrict__ seqP)
{
    const int t = blockIdx.x, h = threadIdx.x;
    if (blockIdx.y == 0) {
        float s = 0.0f;
        for (int l = 0; l < 24; ++l) s += dEmb[(size_t)dCodes[t * 24 + l] * H + h];
        seqD[t * H + h] = s;
    } else {
        float s = 0.0f;
        for (int l = 0; l < 12; ++l) s += pEmb[(size_t)pCodes[t * 12 + l] * H + h];
        seqP[t * H + h] = s;
    }
}

// ---------------------------------------------------------------------------
// GRU recurrence (gi = x@wih^T + bih precomputed by WMMA GEMM).
// grid 2 (block 0 = diag, block 1 = proc), block 256. h lives in LDS.
// ---------------------------------------------------------------------------
__global__ void gru2(const float* __restrict__ giD, const float* __restrict__ whhD,
                     const float* __restrict__ bhhD,
                     const float* __restrict__ giP, const float* __restrict__ whhP,
                     const float* __restrict__ bhhP,
                     float* __restrict__ feats)
{
    const int which = blockIdx.x;
    const float* gi  = which ? giP  : giD;
    const float* whh = which ? whhP : whhD;
    const float* bhh = which ? bhhP : bhhD;
    const int j = threadIdx.x;

    __shared__ float h[H];
    h[j] = 0.0f;
    __syncthreads();

    for (int t = 0; t < T; ++t) {
        const float* gir = gi + t * (3 * H);
        float hr = bhh[j], hz = bhh[H + j], hn = bhh[2 * H + j];
        const float* wr = whh + (size_t)j * H;
        const float* wz = whh + (size_t)(H + j) * H;
        const float* wn = whh + (size_t)(2 * H + j) * H;
        for (int k = 0; k < H; ++k) {
            const float hk = h[k];
            hr = fmaf(wr[k], hk, hr);
            hz = fmaf(wz[k], hk, hz);
            hn = fmaf(wn[k], hk, hn);
        }
        const float r  = 1.0f / (1.0f + expf(-(gir[j] + hr)));
        const float z  = 1.0f / (1.0f + expf(-(gir[H + j] + hz)));
        const float n  = tanhf(gir[2 * H + j] + r * hn);
        const float h2 = (1.0f - z) * n + z * h[j];
        __syncthreads();
        h[j] = h2;
        feats[t * (2 * H) + which * H + j] = h2 > 0.0f ? h2 : 0.0f;
        __syncthreads();
    }
}

// ---------------------------------------------------------------------------
__global__ void combine_prior(const float* __restrict__ pe, const float* __restrict__ pd,
                              const float* __restrict__ b2e, const float* __restrict__ b2d,
                              const float* __restrict__ inter, float* __restrict__ prior)
{
    const int m = blockIdx.x, n = threadIdx.x;
    const size_t idx = (size_t)m * H + n;
    prior[idx] = (pe[idx] + b2e[n]) - inter[0] * (pd[idx] + b2d[n]);
}

// ---------------------------------------------------------------------------
// w_emb = softmax(query @ prior^T) over 4096. One block of 1024 threads.
// ---------------------------------------------------------------------------
__global__ void attn_prior(const float* __restrict__ queries,
                           const float* __restrict__ prior,
                           float* __restrict__ wEmb)
{
    __shared__ float q[H];
    __shared__ float red[32];
    const int tid = threadIdx.x;
    if (tid < H) q[tid] = queries[(T - 1) * H + tid];
    __syncthreads();

    float s[4];
#pragma unroll
    for (int i = 0; i < 4; ++i) {
        const float* pr = prior + (size_t)(tid + i * 1024) * H;
        float acc = 0.0f;
        for (int k = 0; k < H; ++k) acc = fmaf(q[k], pr[k], acc);
        s[i] = acc;
    }
    float lm = fmaxf(fmaxf(s[0], s[1]), fmaxf(s[2], s[3]));
    for (int off = 16; off; off >>= 1) lm = fmaxf(lm, __shfl_down(lm, off, 32));
    if ((tid & 31) == 0) red[tid >> 5] = lm;
    __syncthreads();
    if (tid < 32) {
        float v = red[tid];
        for (int off = 16; off; off >>= 1) v = fmaxf(v, __shfl_down(v, off, 32));
        if (tid == 0) red[0] = v;
    }
    __syncthreads();
    const float gmax = red[0];
    __syncthreads();

    float le = 0.0f;
#pragma unroll
    for (int i = 0; i < 4; ++i) { s[i] = expf(s[i] - gmax); le += s[i]; }
    for (int off = 16; off; off >>= 1) le += __shfl_down(le, off, 32);
    if ((tid & 31) == 0) red[tid >> 5] = le;
    __syncthreads();
    if (tid < 32) {
        float v = red[tid];
        for (int off = 16; off; off >>= 1) v += __shfl_down(v, off, 32);
        if (tid == 0) red[0] = v;
    }
    __syncthreads();
    const float inv = 1.0f / red[0];
#pragma unroll
    for (int i = 0; i < 4; ++i) wEmb[tid + i * 1024] = s[i] * inv;
}

// ---------------------------------------------------------------------------
// visit_w = softmax(query @ history_keys^T) over 31. One block of 32 threads.
// ---------------------------------------------------------------------------
__global__ void attn_hist(const float* __restrict__ queries, float* __restrict__ visitW)
{
    const int t = threadIdx.x;
    __shared__ float sc[32];
    __shared__ float ex[32];
    const float* q = queries + (T - 1) * H;
    float s = -1e30f;
    if (t < T - 1) {
        const float* kt = queries + t * H;
        float acc = 0.0f;
        for (int k = 0; k < H; ++k) acc = fmaf(q[k], kt[k], acc);
        s = acc;
    }
    sc[t] = s;
    __syncthreads();
    float m = -1e30f;
    for (int i = 0; i < T - 1; ++i) m = fmaxf(m, sc[i]);
    const float e = (t < T - 1) ? expf(s - m) : 0.0f;
    ex[t] = e;
    __syncthreads();
    float sum = 0.0f;
    for (int i = 0; i < T - 1; ++i) sum += ex[i];
    if (t < T - 1) visitW[t] = e / sum;
}

// ---------------------------------------------------------------------------
// wv[j] = sum_t visit_w[t] * [j in med_codes[t]]
// ---------------------------------------------------------------------------
__global__ void wv_kernel(const float* __restrict__ visitW,
                          const int* __restrict__ med, float* __restrict__ wv)
{
    const int j = blockIdx.x * 256 + threadIdx.x;
    float acc = 0.0f;
    for (int t = 0; t < T - 1; ++t) {
        bool hit = false;
        for (int l = 0; l < 20; ++l) hit |= (med[t * 20 + l] == j);
        if (hit) acc += visitW[t];
    }
    wv[j] = acc;
}

// ---------------------------------------------------------------------------
// out[m,n] = w_emb[m]*U[m,n] + wv[m]*V[m,n] + proj_b[n]
// ---------------------------------------------------------------------------
__global__ void final_out(const float* __restrict__ U, const float* __restrict__ V,
                          const float* __restrict__ wEmb, const float* __restrict__ wv,
                          const float* __restrict__ pb, float* __restrict__ out)
{
    const int m = blockIdx.x, n = threadIdx.x;
    const size_t idx = (size_t)m * H + n;
    out[idx] = wEmb[m] * U[idx] + wv[m] * V[idx] + pb[n];
}

// ---------------------------------------------------------------------------
extern "C" void kernel_launch(void* const* d_in, const int* in_sizes, int n_in,
                              void* d_out, int out_size, void* d_ws, size_t ws_size,
                              hipStream_t stream)
{
    (void)in_sizes; (void)n_in; (void)out_size; (void)ws_size;

    const int*   diag_codes = (const int*)  d_in[0];
    const int*   proc_codes = (const int*)  d_in[1];
    const int*   med_codes  = (const int*)  d_in[2];
    const float* diag_emb   = (const float*)d_in[3];
    const float* proc_emb   = (const float*)d_in[4];
    const float* gd_wih     = (const float*)d_in[5];
    const float* gd_whh     = (const float*)d_in[6];
    const float* gd_bih     = (const float*)d_in[7];
    const float* gd_bhh     = (const float*)d_in[8];
    const float* gp_wih     = (const float*)d_in[9];
    const float* gp_whh     = (const float*)d_in[10];
    const float* gp_bih     = (const float*)d_in[11];
    const float* gp_bhh     = (const float*)d_in[12];
    const float* q_w        = (const float*)d_in[13];
    const float* q_b        = (const float*)d_in[14];
    const float* adj_ehr    = (const float*)d_in[15];
    const float* adj_ddi    = (const float*)d_in[16];
    const float* w1_ehr     = (const float*)d_in[17];
    const float* b1_ehr     = (const float*)d_in[18];
    const float* w2_ehr     = (const float*)d_in[19];
    const float* b2_ehr     = (const float*)d_in[20];
    const float* w1_ddi     = (const float*)d_in[21];
    const float* b1_ddi     = (const float*)d_in[22];
    const float* w2_ddi     = (const float*)d_in[23];
    const float* b2_ddi     = (const float*)d_in[24];
    const float* inter      = (const float*)d_in[25];
    const float* proj_w     = (const float*)d_in[26];
    const float* proj_b     = (const float*)d_in[27];
    float* out = (float*)d_out;

    // workspace layout (floats)
    float* ws      = (float*)d_ws;
    float* seqD    = ws;                 //  8192
    float* seqP    = seqD   + 8192;      //  8192
    float* giD     = seqP   + 8192;      // 24576
    float* giP     = giD    + 24576;     // 24576
    float* feats   = giP    + 24576;     // 16384
    float* queries = feats  + 16384;     //  8192
    float* wEmb    = queries+ 8192;      //  4096
    float* wv      = wEmb   + 4096;      //  4096
    float* visitW  = wv     + 4096;      //    32
    float* bufA    = ws + 131072;        // 4096*256
    float* bufB    = bufA + (size_t)MVOC * H;
    float* bufP    = bufB + (size_t)MVOC * H;
    _Float16* packBuf = (_Float16*)(bufP + (size_t)MVOC * H);  // up to 4096*256 halves

    auto gemm = [&](const float* A, const float* B, const float* bias, float* C,
                    int M, int N, int K, int lda, int ldb, int ldc,
                    int transB, int relu) {
        pack_b<<<(K * N) / 256, 256, 0, stream>>>(B, packBuf, N, ldb, transB);
        wmma_gemm_pk<<<dim3(M / 16, N / 256), 128, 0, stream>>>(
            A, packBuf, bias, C, M, N, K, lda, ldc, relu);
    };

    // 1) embedding sums
    embed_sum<<<dim3(T, 2), 256, 0, stream>>>(diag_codes, proc_codes,
                                              diag_emb, proc_emb, seqD, seqP);

    // 2) GRU input projections: gi = seq @ wih^T + bih  (M=32, N=768, K=256)
    gemm(seqD, gd_wih, gd_bih, giD, T, 3 * H, H, H, H, 3 * H, 1, 0);
    gemm(seqP, gp_wih, gp_bih, giP, T, 3 * H, H, H, H, 3 * H, 1, 0);

    // 3) GRU recurrences -> feats = relu([od, op])
    gru2<<<2, 256, 0, stream>>>(giD, gd_whh, gd_bhh, giP, gp_whh, gp_bhh, feats);

    // 4) queries = feats @ q_w + q_b  (M=32, N=256, K=512)
    gemm(feats, q_w, q_b, queries, T, H, 2 * H, 2 * H, H, H, 0, 0);

    // 5) GCN(ehr): h1 = relu(adj@w1+b1); t = h1@w2; pe = adj@t
    gemm(adj_ehr, w1_ehr, b1_ehr, bufA, MVOC, H, MVOC, MVOC, H, H, 0, 1);
    gemm(bufA,    w2_ehr, nullptr, bufB, MVOC, H, H,    H,    H, H, 0, 0);
    gemm(adj_ehr, bufB,   nullptr, bufP, MVOC, H, MVOC, MVOC, H, H, 0, 0);

    // 6) GCN(ddi): h1 -> bufA; t -> bufB; pd -> bufA
    gemm(adj_ddi, w1_ddi, b1_ddi, bufA, MVOC, H, MVOC, MVOC, H, H, 0, 1);
    gemm(bufA,    w2_ddi, nullptr, bufB, MVOC, H, H,    H,    H, H, 0, 0);
    gemm(adj_ddi, bufB,   nullptr, bufA, MVOC, H, MVOC, MVOC, H, H, 0, 0);

    // 7) prior = (pe + b2e) - inter*(pd + b2d)   (in bufP)
    combine_prior<<<MVOC, 256, 0, stream>>>(bufP, bufA, b2_ehr, b2_ddi, inter, bufP);

    // 8) attention weights
    attn_prior<<<1, 1024, 0, stream>>>(queries, bufP, wEmb);
    attn_hist<<<1, 32, 0, stream>>>(queries, visitW);
    wv_kernel<<<16, 256, 0, stream>>>(visitW, med_codes, wv);

    // 9) U = prior @ proj_w[:256]; V = prior @ proj_w[256:]
    gemm(bufP, proj_w,                  nullptr, bufA, MVOC, H, H, H, H, H, 0, 0);
    gemm(bufP, proj_w + (size_t)H * H,  nullptr, bufB, MVOC, H, H, H, H, H, 0, 0);

    // 10) out = wEmb.*U + wv.*V + proj_b
    final_out<<<MVOC, 256, 0, stream>>>(bufA, bufB, wEmb, wv, proj_b, out);
}
